// NestedNarx_48687749267795
// MI455X (gfx1250) — compile-verified, matrix operand order checked by reading, not run on hardware
//
#include <hip/hip_runtime.h>
#include <math.h>
#include <float.h>

// ---------------- problem constants ----------------
#define NTT   4096
#define NBAS  2048
#define NXC   8
#define HDIM  64

typedef __attribute__((ext_vector_type(16))) _Float16 v16h;
typedef __attribute__((ext_vector_type(8)))  _Float16 v8h;
typedef __attribute__((ext_vector_type(8)))  float    v8f;

// workspace layout (dword offsets)
//  [0,1024)    : layer1 B fragments, 4 n-tiles (folded 24->16 features; K=16 slot
//                carries b_in so the bias rides the constant-1 feature; K padded to 32)
//  [1024,3072) : layer2 B fragments, 4 n-tiles x 2 K-halves
//  [3072,3584) : layer3 B fragments, 2 K-halves (only n==0 column nonzero)
//  [3648,3712) : b_ih + b_hh (fused)
//  [3712]      : b_out
#define WS_W1    0
#define WS_WIH   1024
#define WS_WOUT  3072
#define WS_BIHH  3648
#define WS_BOUT  3712

#define WPB   4     // waves per block
#define TITER 16    // t-steps per wave-tile (amortizes weight-fragment loads)
#define LROW  72    // padded LDS row stride in halfs (16B-aligned, bank-conflict free)

// ---------------- helpers ----------------

// 16-bit A/B fragment K decomposition (per CDNA5 ISA 7.12.2):
// lane group g = lane>>4; vgpr v<4 holds K = 8g+2v,+1 ; v>=4 holds K = 16+8g+2(v-4),+1
__device__ __host__ inline int kbase_of(int v, int g) {
    return (v < 4) ? (8 * g + 2 * v) : (16 + 8 * g + 2 * (v - 4));
}

// Folded first-layer weights: the 24 raw features contain duplicates
// (EX_OFF = FB_OFF = [-1,-2,-1]) -> 16 effective features:
//   z[0..6]=x[t-1,0..6], z[7..13]=x[t-2,0..6], z[14]=fb(t-1), z[15]=fb(t-2)
__device__ inline float w1eff(const float* Win, int h, int k) {
    if (k < 7)   return Win[h * 24 + k] + Win[h * 24 + 14 + k];
    if (k < 14)  return Win[h * 24 + k];
    if (k == 14) return Win[h * 24 + 21] + Win[h * 24 + 23];
    if (k == 15) return Win[h * 24 + 22];
    return 0.0f; // K padding 17..31 (K=16 handled separately: bias feature)
}

__device__ inline float fast_tanh(float v) {
#if __has_builtin(__builtin_amdgcn_tanhf)
    return __builtin_amdgcn_tanhf(v);   // v_tanh_f32 (CDNA5 trans op)
#else
    return tanhf(v);
#endif
}

__device__ inline float relu1(float v) {
#if __has_builtin(__builtin_amdgcn_fmed3f)
    return __builtin_amdgcn_fmed3f(v, 0.0f, FLT_MAX);  // single v_med3_f32
#else
    return fmaxf(v, 0.0f);
#endif
}

// upstream feedback: sum of previously-computed output columns
__device__ inline float upsum(const float* out, int t, int j,
                              int upA, int upB, int upBjmax) {
    float s = 0.0f;
    if (upA >= 0) s += out[(size_t)t * NBAS + upA + j];
    if (upB >= 0 && j < upBjmax) s += out[(size_t)t * NBAS + upB + j];
    return s;
}

// ---------------- weight prep: pack B fragments in lane order ----------------
__global__ void narx_prep_kernel(const float* __restrict__ Win,
                                 const float* __restrict__ b_in,
                                 const float* __restrict__ Wih,
                                 const float* __restrict__ b_ih,
                                 const float* __restrict__ b_hh,
                                 const float* __restrict__ Wout,
                                 const float* __restrict__ b_out,
                                 float* __restrict__ ws) {
    int idx = blockIdx.x * blockDim.x + threadIdx.x;
    _Float16* wsH = (_Float16*)ws;
    if (idx < 3584) {
        // one packed dword (two f16) of some B fragment
        int r = idx & 255, lane = r >> 3, v = r & 7;
        int n = lane & 15, g = lane >> 4;
        int kb = kbase_of(v, g);
        float lo, hi;
        if (idx < 1024) {                       // layer1: W1eff + bias at K=16
            int nt = idx >> 8;
            int h  = nt * 16 + n;
            lo = (kb == 16) ? b_in[h] : w1eff(Win, h, kb);   // kb even, so kb+1 != 16
            hi = w1eff(Win, h, kb + 1);
        } else if (idx < 3072) {                // layer2: Wih[h_out][h_in]
            int fi = (idx - 1024) >> 8;
            int nt = fi >> 1, kh = fi & 1;
            int row = (nt * 16 + n) * HDIM + kh * 32;
            lo = Wih[row + kb];
            hi = Wih[row + kb + 1];
        } else {                                // layer3: Wout column 0 only
            int kh = (idx - 3072) >> 8;
            lo = (n == 0) ? Wout[kh * 32 + kb] : 0.0f;
            hi = (n == 0) ? Wout[kh * 32 + kb + 1] : 0.0f;
        }
        wsH[idx * 2]     = (_Float16)lo;
        wsH[idx * 2 + 1] = (_Float16)hi;
    } else if (idx >= 3648 && idx < 3712) {
        ws[idx] = b_ih[idx - 3648] + b_hh[idx - 3648];
    } else if (idx == 3712) {
        ws[idx] = b_out[0];
    }
}

// ---------------- warmup rows t=0,1 : out = x[...,7] + upstream ----------------
__global__ void narx_warmup_kernel(const float* __restrict__ x, float* out,
                                   int nbg, int bbase, int colbase,
                                   int upA, int upB, int upBjmax) {
    int idx = blockIdx.x * blockDim.x + threadIdx.x;
    if (idx >= 2 * nbg) return;
    int t = idx / nbg, j = idx % nbg;
    out[(size_t)t * NBAS + colbase + j] =
        x[((size_t)t * NBAS + (bbase + j)) * NXC + 7] + upsum(out, t, j, upA, upB, upBjmax);
}

// ---------------- main NARX pass: one wave = 16 basins x TITER timesteps ----------------
__global__ __launch_bounds__(32 * WPB) void narx_main_kernel(
        const float* __restrict__ x, const float* __restrict__ ws,
        float* out, int nbt, int bbase, int colbase,
        int upA, int upB, int upBjmax) {
    __shared__ __attribute__((aligned(16))) _Float16 xlds[WPB][16 * LROW];
    __shared__ float ystage[WPB][16];

    const int w    = threadIdx.x >> 5;
    const int lane = threadIdx.x & 31;
    const int n    = lane & 15;          // row (M) for A; column (N) for B/C/D
    const int g    = lane >> 4;          // lane half -> K group / M offset in C
    const int ttiles = (NTT - 2 + TITER - 1) / TITER;   // 256
    const int total  = nbt * ttiles;

    int tile = blockIdx.x * WPB + w;
    if (tile >= total) return;           // wave-uniform exit
    const int bt = tile % nbt;
    const int tt = tile / nbt;
    const int b  = bbase + bt * 16 + n;  // this lane's basin row
    const int j  = bt * 16 + n;          // basin index within group
    const int col0 = colbase + bt * 16;

    // ---- per-wave constant weight fragments (one contiguous 32B load per frag/lane)
    const v16h* fr = (const v16h*)ws;
    v16h b1[4], b2[8], b3[2];
    for (int i = 0; i < 4; ++i) b1[i] = fr[(0  + i) * 32 + lane];
    for (int i = 0; i < 8; ++i) b2[i] = fr[(4  + i) * 32 + lane];
    for (int i = 0; i < 2; ++i) b3[i] = fr[(12 + i) * 32 + lane];

    // ---- loop-invariant bias-seeded accumulators (C operand of WMMA):
    // C/D layout puts column h on lane (n = h%16), so a lane-splat is bias[h].
    v8f cb2[4], cb3;
    for (int nt = 0; nt < 4; ++nt) {
        float bv = ws[WS_BIHH + nt * 16 + n];
        for (int i = 0; i < 8; ++i) cb2[nt][i] = bv;
    }
    {
        float bv = ws[WS_BOUT];
        for (int i = 0; i < 8; ++i) cb3[i] = bv;
    }
    const _Float16 onef = (g == 0) ? (_Float16)1.0f : (_Float16)0.0f; // K=16 bias feature
    const int mbase = 8 * g;             // C/D tile: vgpr v holds M = v + 8*(lane>=16)
    const _Float16* lr = &xlds[w][n * LROW + 8 * g];

    const int t0   = 2 + tt * TITER;
    const int tend = (t0 + TITER < NTT) ? (t0 + TITER) : NTT;

    for (int t = t0; t < tend; ++t) {
        // ---- gather: lanes 0-15 read x[t-1,b,0..7], lanes 16-31 read x[t-2,b,0..7]
        const float* xp = x + ((size_t)(t - 1 - g) * NBAS + b) * NXC;
        float4 o0 = *(const float4*)xp;
        float4 o1 = *(const float4*)(xp + 4);
        __builtin_prefetch(xp + (size_t)NBAS * NXC, 0, 1);  // next t row (speculative)
        float othc0 = __shfl_xor(o0.x, 16);   // x[t-2,b,0] for g0 lanes
        float othc7 = __shfl_xor(o1.w, 16);   // x[t-1,b,7] for g1 lanes
        float up1 = 0.0f, up2 = 0.0f;
        if (g) {                              // feedback taps only needed in K=8..15 half
            up1 = upsum(out, t - 1, j, upA, upB, upBjmax);
            up2 = upsum(out, t - 2, j, upA, upB, upBjmax);
        }
        // ---- layer1 A fragment in-register (K 0..15 features, K=16 constant-1 bias)
        _Float16 z0, z1, z2, z3, z4, z5, z6, z7;
        if (g == 0) {
            z0 = (_Float16)o0.x; z1 = (_Float16)o0.y; z2 = (_Float16)o0.z; z3 = (_Float16)o0.w;
            z4 = (_Float16)o1.x; z5 = (_Float16)o1.y; z6 = (_Float16)o1.z; z7 = (_Float16)othc0;
        } else {
            z0 = (_Float16)o0.y; z1 = (_Float16)o0.z; z2 = (_Float16)o0.w; z3 = (_Float16)o1.x;
            z4 = (_Float16)o1.y; z5 = (_Float16)o1.z;
            z6 = (_Float16)(othc7 + up1);     // fb(t-1)
            z7 = (_Float16)(o1.w + up2);      // fb(t-2)
        }
        v16h a1 = { z0, z1, z2, z3, z4, z5, z6, z7,
                    onef, (_Float16)0, (_Float16)0, (_Float16)0,
                    (_Float16)0, (_Float16)0, (_Float16)0, (_Float16)0 };

        // ---- layer1: xl = relu(z @ W1eff^T + b_in), 4 WMMAs (bias inside WMMA)
        v8f c1[4];
        for (int nt = 0; nt < 4; ++nt) {
            v8f zero = {};
            c1[nt] = __builtin_amdgcn_wmma_f32_16x16x32_f16(
                         false, a1, false, b1[nt], (short)0, zero, false, false);
        }
        for (int nt = 0; nt < 4; ++nt) {
            const int hh = nt * 16 + n;
            for (int v = 0; v < 8; ++v) {
                float a = relu1(c1[nt][v]);               // v_med3_f32
                xlds[w][(mbase + v) * LROW + hh] = (_Float16)a;
            }
        }
        __builtin_amdgcn_wave_barrier();
        asm volatile("s_wait_dscnt 0" ::: "memory");

        // ---- transpose: reload xl as A fragments (2 x v16h via ds_load_b128)
        v8h l0 = *(const v8h*)(lr);
        v8h l1 = *(const v8h*)(lr + 16);
        v8h l2 = *(const v8h*)(lr + 32);
        v8h l3 = *(const v8h*)(lr + 48);
        v16h a2k0 = __builtin_shufflevector(l0, l1, 0,1,2,3,4,5,6,7,8,9,10,11,12,13,14,15);
        v16h a2k1 = __builtin_shufflevector(l2, l3, 0,1,2,3,4,5,6,7,8,9,10,11,12,13,14,15);

        // ---- layer2: h = tanh(xl @ Wih^T + bias), 8 WMMAs, bias via seeded C
        v8f c2[4];
        for (int nt = 0; nt < 4; ++nt) {
            v8f p = __builtin_amdgcn_wmma_f32_16x16x32_f16(
                        false, a2k0, false, b2[nt * 2], (short)0, cb2[nt], false, false);
            c2[nt] = __builtin_amdgcn_wmma_f32_16x16x32_f16(
                        false, a2k1, false, b2[nt * 2 + 1], (short)0, p, false, false);
        }
        __builtin_amdgcn_wave_barrier();
        for (int nt = 0; nt < 4; ++nt) {
            const int hh = nt * 16 + n;
            for (int v = 0; v < 8; ++v) {
                float a = fast_tanh(c2[nt][v]);           // v_tanh_f32
                xlds[w][(mbase + v) * LROW + hh] = (_Float16)a;
            }
        }
        __builtin_amdgcn_wave_barrier();
        asm volatile("s_wait_dscnt 0" ::: "memory");

        // ---- layer3: y = h @ Wout^T + b_out (Wout padded to 16 cols, col 0 live)
        v8h q0 = *(const v8h*)(lr);
        v8h q1 = *(const v8h*)(lr + 16);
        v8h q2 = *(const v8h*)(lr + 32);
        v8h q3 = *(const v8h*)(lr + 48);
        v16h a3k0 = __builtin_shufflevector(q0, q1, 0,1,2,3,4,5,6,7,8,9,10,11,12,13,14,15);
        v16h a3k1 = __builtin_shufflevector(q2, q3, 0,1,2,3,4,5,6,7,8,9,10,11,12,13,14,15);
        v8f cy = __builtin_amdgcn_wmma_f32_16x16x32_f16(
                     false, a3k0, false, b3[0], (short)0, cb3, false, false);
        cy = __builtin_amdgcn_wmma_f32_16x16x32_f16(
                     false, a3k1, false, b3[1], (short)0, cy, false, false);

        // column 0 of D lives in lanes 0 (M=0..7) and 16 (M=8..15)
        if (n == 0) {
            for (int v = 0; v < 8; ++v) ystage[w][mbase + v] = cy[v];
        }
        __builtin_amdgcn_wave_barrier();
        asm volatile("s_wait_dscnt 0" ::: "memory");
        if (lane < 16) out[(size_t)t * NBAS + col0 + lane] = ystage[w][lane];
    }
}

// ---------------- host-side launcher ----------------
extern "C" void kernel_launch(void* const* d_in, const int* in_sizes, int n_in,
                              void* d_out, int out_size, void* d_ws, size_t ws_size,
                              hipStream_t stream) {
    (void)in_sizes; (void)n_in; (void)out_size; (void)ws_size;
    const float* x     = (const float*)d_in[0];
    const float* Win   = (const float*)d_in[1];
    const float* b_in  = (const float*)d_in[2];
    const float* Wih   = (const float*)d_in[3];
    const float* b_ih  = (const float*)d_in[4];
    const float* b_hh  = (const float*)d_in[5];
    const float* Wout  = (const float*)d_in[6];
    const float* b_out = (const float*)d_in[7];
    float* out = (float*)d_out;
    float* ws  = (float*)d_ws;

    narx_prep_kernel<<<15, 256, 0, stream>>>(Win, b_in, Wih, b_ih, b_hh, Wout, b_out, ws);

    // pass order: highest basin order first; feedback read from columns written
    // by the previous pass (stream-ordered). {nbg, bbase, colbase, upA, upB, upBjmax}
    struct G { int nbg, bbase, colbase, upA, upB, upBjmax; };
    const G gs[4] = {
        { 768, 1280,    0,   -1,   -1,   0 },   // order 3 (headwaters)
        { 512,  768,  768,    0,  512, 256 },   // order 2: cols j and j+512 (j<256)
        { 512,  256, 1280,  768,   -1,   0 },   // order 1: col 768+j
        { 256,    0, 1792, 1280, 1536, 256 },   // order 0: cols 1280+j and 1536+j
    };
    for (int i = 0; i < 4; ++i) {
        const G& q = gs[i];
        const int nbt    = q.nbg / 16;
        const int wtiles = nbt * ((NTT - 2 + TITER - 1) / TITER);
        const int blocks = (wtiles + WPB - 1) / WPB;
        narx_warmup_kernel<<<(2 * q.nbg + 255) / 256, 256, 0, stream>>>(
            x, out, q.nbg, q.bbase, q.colbase, q.upA, q.upB, q.upBjmax);
        narx_main_kernel<<<blocks, 32 * WPB, 0, stream>>>(
            x, ws, out, nbt, q.bbase, q.colbase, q.upA, q.upB, q.upBjmax);
    }
}
